// QuantizedLayerNorm_73684458930260
// MI455X (gfx1250) — compile-verified, matrix operand order checked by reading, not run on hardware
//
#include <hip/hip_runtime.h>

// Quantized LayerNorm for MI455X (gfx1250, wave32).
// Memory-bound: 512 MB total traffic -> ~22 us floor @ 23.3 TB/s.
// Pass 1 streams f32 x, converts to fp16 (the reference's q_input), stages the
// fp16 tile in LDS (32 KB/wave), and reduces row sum / sum-of-squares with
// v_wmma_f32_16x16x32_f16 (A = ones, dual accumulators to avoid WMMA->WMMA
// hazard NOPs). Pass 2 re-reads fp16 from LDS and applies pre-quantized fp16
// weight/bias (prepared once into d_ws) via mixed-precision FMAs.

typedef __attribute__((ext_vector_type(16))) _Float16 v16h;
typedef __attribute__((ext_vector_type(8)))  float    v8f;

#define N_COLS          1024
#define CHUNK           32                 // K per WMMA
#define NCHUNK          (N_COLS / CHUNK)   // 32
#define ROWS_PER_WAVE   16
#define WAVES_PER_BLOCK 2
#define ROWS_PER_BLOCK  (ROWS_PER_WAVE * WAVES_PER_BLOCK)
#define PF_DIST         (8 * CHUNK)        // prefetch 1 KB ahead

// fp16 quantization with saturation (matches _fp_quant: RNE, clamp at 65504,
// fp16 subnormals preserved since denorm mode is "allow").
__device__ __forceinline__ float q16(float v) {
    v = fminf(fmaxf(v, -65504.0f), 65504.0f);
    return (float)(_Float16)v;
}

__device__ __forceinline__ v16h pack16(const float4 a0, const float4 a1,
                                       const float4 a2, const float4 a3) {
    v16h v;
    v[0]  = (_Float16)a0.x; v[1]  = (_Float16)a0.y;
    v[2]  = (_Float16)a0.z; v[3]  = (_Float16)a0.w;
    v[4]  = (_Float16)a1.x; v[5]  = (_Float16)a1.y;
    v[6]  = (_Float16)a1.z; v[7]  = (_Float16)a1.w;
    v[8]  = (_Float16)a2.x; v[9]  = (_Float16)a2.y;
    v[10] = (_Float16)a2.z; v[11] = (_Float16)a2.w;
    v[12] = (_Float16)a3.x; v[13] = (_Float16)a3.y;
    v[14] = (_Float16)a3.z; v[15] = (_Float16)a3.w;
    return v;
}

// One-time prep: quantize weight/bias to fp16 into workspace.
__global__ void quant_wb_kernel(const float* __restrict__ w,
                                const float* __restrict__ b,
                                _Float16* __restrict__ qw,
                                _Float16* __restrict__ qb, int n) {
    const int i = blockIdx.x * blockDim.x + threadIdx.x;
    if (i < n) {
        qw[i] = (_Float16)fminf(fmaxf(w[i], -65504.0f), 65504.0f);
        qb[i] = (_Float16)fminf(fmaxf(b[i], -65504.0f), 65504.0f);
    }
}

__global__ __launch_bounds__(WAVES_PER_BLOCK * 32, 1)
void qln_wmma_kernel(const float* __restrict__ x,
                     const _Float16* __restrict__ qw,
                     const _Float16* __restrict__ qb,
                     float* __restrict__ out,
                     int rows)
{
    // fp16 staging: [wave][chunk][lane] -> 32 B per lane per chunk. 64 KB total.
    __shared__ v16h stage[WAVES_PER_BLOCK][NCHUNK][32];

    const int tid  = threadIdx.x;
    const int wave = tid >> 5;
    const int lane = tid & 31;
    const int r    = lane & 15;   // which row of the 16-row tile this lane owns
    const int h    = lane >> 4;   // which 16-wide half of each 32-wide K chunk

    const int rowBase = blockIdx.x * ROWS_PER_BLOCK + wave * ROWS_PER_WAVE;
    if (rowBase >= rows) return;  // wave-uniform guard (EXEC all-1s for WMMA)

    const int row = rowBase + r;
    const float* __restrict__ p = x + (size_t)row * N_COLS;

    // A-matrix = all ones (layout-independent): D[m][n] = sum_k B[k][n].
    v16h ones;
#pragma unroll
    for (int j = 0; j < 16; ++j) ones[j] = (_Float16)1.0f;

    // Dual accumulator chains: consecutive WMMAs never share D/C registers,
    // eliminating the 5-slot WMMA->WMMA RAW hazard NOPs.
    v8f accS0 = {}, accS1 = {};
    v8f accQ0 = {}, accQ1 = {};

    // ---- Pass 1: stream x from HBM, quantize to fp16, stage to LDS, reduce ----
#pragma unroll 2
    for (int c = 0; c < NCHUNK; c += 2) {
#pragma unroll
        for (int u = 0; u < 2; ++u) {
            const int cc   = c + u;
            const int col0 = cc * CHUNK + h * 16;  // lane reads 64 contiguous B
            const float4 a0 = *(const float4*)(p + col0 + 0);
            const float4 a1 = *(const float4*)(p + col0 + 4);
            const float4 a2 = *(const float4*)(p + col0 + 8);
            const float4 a3 = *(const float4*)(p + col0 + 12);

            // gfx1250 prefetch well ahead of the load stream
            __builtin_prefetch(p + col0 + PF_DIST, 0, 1);

            const v16h bvec = pack16(a0, a1, a2, a3); // q_input chunk (B layout)
            const v16h bsq  = bvec * bvec;            // v_pk_mul_f16

            stage[wave][cc][lane] = bvec;             // 2x ds_store_b128

            if (u == 0) {
                accS0 = __builtin_amdgcn_wmma_f32_16x16x32_f16(
                            false, ones, false, bvec, (short)0, accS0, false, false);
                accQ0 = __builtin_amdgcn_wmma_f32_16x16x32_f16(
                            false, ones, false, bsq,  (short)0, accQ0, false, false);
            } else {
                accS1 = __builtin_amdgcn_wmma_f32_16x16x32_f16(
                            false, ones, false, bvec, (short)0, accS1, false, false);
                accQ1 = __builtin_amdgcn_wmma_f32_16x16x32_f16(
                            false, ones, false, bsq,  (short)0, accQ1, false, false);
            }
        }
    }

    // D VGPR0, lane l holds column n = l%16 == this lane's row.
    const float inv   = 1.0f / (float)N_COLS;
    const float mean  = (accS0[0] + accS1[0]) * inv;
    const float var   = (accQ0[0] + accQ1[0]) * inv - mean * mean;
    const float qmean = q16(mean);
    const float qvar  = q16(var);
    const float rstd  = 1.0f / sqrtf(qvar + 1e-5f);
    const float mr    = qmean * rstd;   // out = hw*(hv*rstd - mr) + hb

    // ---- Pass 2: re-read fp16 tile from LDS, normalize, write f32 ----
    float* __restrict__ q = out + (size_t)row * N_COLS;
#pragma unroll 4
    for (int c = 0; c < NCHUNK; ++c) {
        const int col0 = c * CHUNK + h * 16;
        const v16h hv = stage[wave][c][lane];        // 2x ds_load_b128
        const v16h hw = *(const v16h*)(qw + col0);   // 16 halves = 32 B
        const v16h hb = *(const v16h*)(qb + col0);

        // (float)f16 * f32 +/- ... -> v_fma_mix_f32 candidates
#define QLN_OUT(J) ((float)hw[J] * ((float)hv[J] * rstd - mr) + (float)hb[J])
        float4 o0, o1, o2, o3;
        o0.x = QLN_OUT(0);  o0.y = QLN_OUT(1);  o0.z = QLN_OUT(2);  o0.w = QLN_OUT(3);
        o1.x = QLN_OUT(4);  o1.y = QLN_OUT(5);  o1.z = QLN_OUT(6);  o1.w = QLN_OUT(7);
        o2.x = QLN_OUT(8);  o2.y = QLN_OUT(9);  o2.z = QLN_OUT(10); o2.w = QLN_OUT(11);
        o3.x = QLN_OUT(12); o3.y = QLN_OUT(13); o3.z = QLN_OUT(14); o3.w = QLN_OUT(15);
#undef QLN_OUT

        *(float4*)(q + col0 + 0)  = o0;   // 4x global_store_b128, 64 B/lane
        *(float4*)(q + col0 + 4)  = o1;
        *(float4*)(q + col0 + 8)  = o2;
        *(float4*)(q + col0 + 12) = o3;
    }
}

extern "C" void kernel_launch(void* const* d_in, const int* in_sizes, int n_in,
                              void* d_out, int out_size, void* d_ws, size_t ws_size,
                              hipStream_t stream) {
    const float* x      = (const float*)d_in[0];
    const float* weight = (const float*)d_in[1];
    const float* bias   = (const float*)d_in[2];
    float* out          = (float*)d_out;

    const int n_cols = N_COLS;                         // 1024
    _Float16* qw = (_Float16*)d_ws;                    // 2 KB
    _Float16* qb = qw + n_cols;                        // 2 KB, 32B-aligned

    const int rows   = in_sizes[0] / N_COLS;           // 65536
    const int blocks = (rows + ROWS_PER_BLOCK - 1) / ROWS_PER_BLOCK;

    hipLaunchKernelGGL(quant_wb_kernel, dim3((n_cols + 255) / 256), dim3(256),
                       0, stream, weight, bias, qw, qb, n_cols);
    hipLaunchKernelGGL(qln_wmma_kernel, dim3(blocks), dim3(WAVES_PER_BLOCK * 32),
                       0, stream, x, qw, qb, out, rows);
}
